// ATT_HGCN_39883066310780
// MI455X (gfx1250) — matrix operand order repaired and checked by reading, not compiled, not run on hardware
//
#include <hip/hip_runtime.h>
#include <hip/hip_bf16.h>

// ---------------------------------------------------------------------------
// Types for CDNA5 WMMA
// ---------------------------------------------------------------------------
typedef __attribute__((ext_vector_type(16))) __bf16 v16bf;
typedef __attribute__((ext_vector_type(8)))  float  v8f;

__device__ __forceinline__ unsigned short f32_to_bf16_rne(float f) {
    union { float f; unsigned u; } v; v.f = f;
    unsigned u = v.u;
    unsigned r = u + 0x7FFFu + ((u >> 16) & 1u);   // round-to-nearest-even
    return (unsigned short)(r >> 16);
}
__device__ __forceinline__ unsigned pack_bf16x2(float lo, float hi) {
    return (unsigned)f32_to_bf16_rne(lo) | ((unsigned)f32_to_bf16_rne(hi) << 16);
}

// ---------------------------------------------------------------------------
// Zero-fill (d_ws / d_out regions are poisoned by the harness)
// ---------------------------------------------------------------------------
__global__ void fill_zero_kernel(float* __restrict__ p, size_t n) {
    size_t i = (size_t)blockIdx.x * blockDim.x + threadIdx.x;
    size_t stride = (size_t)gridDim.x * blockDim.x;
    for (; i < n; i += stride) p[i] = 0.0f;
}

// ---------------------------------------------------------------------------
// Streaming f32 -> bf16 conversion (done ONCE per activation matrix; each
// matrix feeds two GEMMs, so the hot GEMM loop needs no conversion VALU).
// ---------------------------------------------------------------------------
__global__ __launch_bounds__(256)
void cvt_bf16_kernel(const float* __restrict__ in, unsigned short* __restrict__ out,
                     size_t n4 /* n/4 */) {
    size_t i = (size_t)blockIdx.x * blockDim.x + threadIdx.x;
    size_t stride = (size_t)gridDim.x * blockDim.x;
    const float4* in4 = (const float4*)in;
    uint2* out4 = (uint2*)out;
    for (; i < n4; i += stride) {
        float4 v = in4[i];
        uint2 o;
        o.x = pack_bf16x2(v.x, v.y);
        o.y = pack_bf16x2(v.z, v.w);
        out4[i] = o;
    }
}

// ---------------------------------------------------------------------------
// Fold attention projections:  vk = w_k @ a[:64],  vq = w_q @ a[64:]
// ---------------------------------------------------------------------------
__global__ void prep_vkq_kernel(const float* __restrict__ wq,
                                const float* __restrict__ wk,
                                const float* __restrict__ watt,
                                float* __restrict__ vk, float* __restrict__ vq, int D) {
    int d = blockIdx.x * blockDim.x + threadIdx.x;
    if (d >= D) return;
    float sk = 0.f, sq = 0.f;
    for (int a = 0; a < 64; ++a) {
        sk += wk[d * 64 + a] * watt[a];
        sq += wq[d * 64 + a] * watt[64 + a];
    }
    vk[d] = sk;
    vq[d] = sq;
}

// ---------------------------------------------------------------------------
// GEMM  C[M x NCOLS] = A[M x 128] * B[128 x NCOLS]
//   A pre-converted to bf16 (row-major) -> A fragment = 2x global_load_b128.
//   B (f32 weights) staged transposed+bf16 in LDS (16B-aligned rows).
//   NCOLS templated so C-store strides fold into immediate offsets.
//   Block = 8 waves; each wave owns a 16x64 C tile; K-loop fully unrolled.
//   Per K-step: issue all 8 ds_load_b128 into independent registers, then
//   burst 4 v_wmma back-to-back (no per-WMMA dscnt-0 serialization).
// ---------------------------------------------------------------------------
#define BPAD 136   // LDS row stride in ushorts: 272B = 17*16B

template <int NCOLS>
__global__ __launch_bounds__(256)
void gemm_bf16_wmma_kernel(const unsigned short* __restrict__ A16,
                           const float* __restrict__ B,
                           float* __restrict__ C, int M) {
    constexpr int K = 128;
    __shared__ unsigned short ldsB[64 * BPAD];

    const int tid   = threadIdx.x;
    const int lane  = tid & 31;
    const int wv    = tid >> 5;            // 0..7
    const int cbase = blockIdx.y * 64;

    // Cooperative load of B panel (128 x 64 cols), transposed + bf16-converted.
    for (int i = tid; i < K * 64; i += 256) {
        int k = i >> 6, c = i & 63;        // coalesced global read
        ldsB[c * BPAD + k] = f32_to_bf16_rne(B[(size_t)k * NCOLS + cbase + c]);
    }
    __syncthreads();

    const int rt = blockIdx.x * 8 + wv;    // 16-row tile index
    if (rt * 16 >= M) return;              // wave-uniform (M is a multiple of 16)

    const int row   = rt * 16 + (lane & 15);
    const int hhalf = lane >> 4;           // 0/1: which K-half this lane holds
    const unsigned short* arow = A16 + (size_t)row * K;

    v8f acc[4] = {};                       // 4 x (16x16 f32) accumulators

    #pragma unroll
    for (int kk = 0; kk < K; kk += 32) {
        // A fragment (16x32 bf16): elems 0-7 = K kk+h*8.., elems 8-15 = kk+16+h*8..
        union { uint4 q[2]; v16bf v; } af;
        af.q[0] = *(const uint4*)(arow + kk + hhalf * 8);
        af.q[1] = *(const uint4*)(arow + kk + 16 + hhalf * 8);

        // Load ALL 4 column-subtile B fragments into independent registers.
        union { uint4 q[2]; v16bf v; } bf[4];
        #pragma unroll
        for (int ct = 0; ct < 4; ++ct) {
            const unsigned short* bp =
                &ldsB[(ct * 16 + (lane & 15)) * BPAD + kk + hhalf * 8];
            bf[ct].q[0] = *(const uint4*)(bp);       // 16B-aligned ds_load_b128
            bf[ct].q[1] = *(const uint4*)(bp + 16);
        }
        // Burst the 4 WMMAs.
        #pragma unroll
        for (int ct = 0; ct < 4; ++ct) {
            acc[ct] = __builtin_amdgcn_wmma_f32_16x16x32_bf16(
                false, af.v, false, bf[ct].v, (short)0, acc[ct], false, false);
        }
    }

    // Store: element j of each acc is row (hhalf*8 + j); strides are constants.
    float* cp = C + (size_t)(rt * 16 + hhalf * 8) * NCOLS + cbase + (lane & 15);
    #pragma unroll
    for (int ct = 0; ct < 4; ++ct) {
        #pragma unroll
        for (int j = 0; j < 8; ++j) {
            cp[(size_t)j * NCOLS + ct * 16] = acc[ct][j];
        }
    }
}

// ---------------------------------------------------------------------------
// SpMM: out[dst] += w * feat[src]   (one wave per edge; gather stays L2-resident)
// ---------------------------------------------------------------------------
template <int VEC>
__global__ __launch_bounds__(256)
void spmm_kernel(const int* __restrict__ src, const int* __restrict__ dst,
                 const float* __restrict__ ew, const float* __restrict__ feat,
                 float* __restrict__ out, int E) {
    constexpr int D = VEC * 32;
    int e    = (int)((blockIdx.x * (size_t)blockDim.x + threadIdx.x) >> 5);
    int lane = threadIdx.x & 31;
    if (e >= E) return;                    // wave-uniform
    int   s = src[e], d = dst[e];
    float w = ew[e];
    if (VEC == 4) {                        // D == 128
        const float4* f = (const float4*)(feat + (size_t)s * D);
        float4 v = f[lane];
        float* o = out + (size_t)d * D + lane * 4;
        atomicAdd(o + 0, w * v.x); atomicAdd(o + 1, w * v.y);
        atomicAdd(o + 2, w * v.z); atomicAdd(o + 3, w * v.w);
    } else {                               // D == 64
        const float2* f = (const float2*)(feat + (size_t)s * D);
        float2 v = f[lane];
        float* o = out + (size_t)d * D + lane * 2;
        atomicAdd(o + 0, w * v.x); atomicAdd(o + 1, w * v.y);
    }
}

// ---------------------------------------------------------------------------
// Fused attention (one wave per node), D templated -> vector loads, operands
// kept in registers for the output pass.
// ---------------------------------------------------------------------------
__device__ __forceinline__ float elu1(float x) { return x > 0.f ? x : expf(x) - 1.f; }

template <int D>
__global__ __launch_bounds__(256)
void attn_kernel(const float* __restrict__ selfF, const float* __restrict__ nbF,
                 const float* __restrict__ vk, const float* __restrict__ vq,
                 const float* __restrict__ bias, float* __restrict__ outH,
                 float* __restrict__ attOut, int n, int applyElu) {
    int node = (int)((blockIdx.x * (size_t)blockDim.x + threadIdx.x) >> 5);
    int lane = threadIdx.x & 31;
    if (node >= n) return;                 // wave-uniform

    const float* sp = selfF + (size_t)node * D;
    const float* np = nbF   + (size_t)node * D;
    float dSK, dSQ, dNK;
    float4 s4, n4;  float2 s2, n2;

    if (D == 128) {
        s4 = ((const float4*)sp)[lane];
        n4 = ((const float4*)np)[lane];
        float4 k = ((const float4*)vk)[lane];
        float4 q = ((const float4*)vq)[lane];
        dSK = s4.x*k.x + s4.y*k.y + s4.z*k.z + s4.w*k.w;
        dSQ = s4.x*q.x + s4.y*q.y + s4.z*q.z + s4.w*q.w;
        dNK = n4.x*k.x + n4.y*k.y + n4.z*k.z + n4.w*k.w;
    } else {
        s2 = ((const float2*)sp)[lane];
        n2 = ((const float2*)np)[lane];
        float2 k = ((const float2*)vk)[lane];
        float2 q = ((const float2*)vq)[lane];
        dSK = s2.x*k.x + s2.y*k.y;
        dSQ = s2.x*q.x + s2.y*q.y;
        dNK = n2.x*k.x + n2.y*k.y;
    }
    #pragma unroll
    for (int off = 16; off > 0; off >>= 1) {
        dSK += __shfl_xor(dSK, off, 32);
        dSQ += __shfl_xor(dSQ, off, 32);
        dNK += __shfl_xor(dNK, off, 32);
    }
    float eS = elu1(dSK + dSQ);
    float eN = elu1(dNK + dSQ);
    float m  = fmaxf(eS, eN);
    float p0 = expf(eS - m), p1 = expf(eN - m);
    float inv = 1.f / (p0 + p1);
    float a0 = p0 * inv, a1 = p1 * inv;
    if (lane == 0) { attOut[(size_t)node * 2] = a0; attOut[(size_t)node * 2 + 1] = a1; }

    float* op = outH + (size_t)node * D;
    if (D == 128) {
        float4 b = ((const float4*)bias)[lane];
        float4 o;
        o.x = a0*s4.x + a1*n4.x + b.x;  o.y = a0*s4.y + a1*n4.y + b.y;
        o.z = a0*s4.z + a1*n4.z + b.z;  o.w = a0*s4.w + a1*n4.w + b.w;
        if (applyElu) { o.x = elu1(o.x); o.y = elu1(o.y); o.z = elu1(o.z); o.w = elu1(o.w); }
        ((float4*)op)[lane] = o;
    } else {
        float2 b = ((const float2*)bias)[lane];
        float2 o;
        o.x = a0*s2.x + a1*n2.x + b.x;  o.y = a0*s2.y + a1*n2.y + b.y;
        if (applyElu) { o.x = elu1(o.x); o.y = elu1(o.y); }
        ((float2*)op)[lane] = o;
    }
}

// ---------------------------------------------------------------------------
// Graph pooling: LDS-accumulated segment sums (graph_ids sorted, G = 64)
// ---------------------------------------------------------------------------
__global__ __launch_bounds__(256)
void pool_kernel(const float* __restrict__ h2, const int* __restrict__ gid,
                 float* __restrict__ gsum, float* __restrict__ gcnt,
                 int n, int nodesPerBlock) {
    __shared__ float lsum[64 * 64];
    __shared__ float lcnt[64];
    int tid = threadIdx.x;
    for (int i = tid; i < 64 * 64; i += 256) lsum[i] = 0.f;
    if (tid < 64) lcnt[tid] = 0.f;
    __syncthreads();

    int d = tid & 63, sub = tid >> 6;
    int start = blockIdx.x * nodesPerBlock;
    int end   = min(start + nodesPerBlock, n);
    for (int nd = start + sub; nd < end; nd += 4) {
        int g = gid[nd];
        atomicAdd(&lsum[g * 64 + d], h2[(size_t)nd * 64 + d]);
        if (d == 0) atomicAdd(&lcnt[g], 1.f);
    }
    __syncthreads();
    for (int g = sub; g < 64; g += 4) atomicAdd(&gsum[g * 64 + d], lsum[g * 64 + d]);
    if (tid < 64) atomicAdd(&gcnt[tid], lcnt[tid]);
}

// ---------------------------------------------------------------------------
// Final tiny MLP over 64 graphs:  relu(hg @ w1 + b1) @ w2 + b2
// ---------------------------------------------------------------------------
__global__ void mlp_kernel(const float* __restrict__ gsum, const float* __restrict__ gcnt,
                           const float* __restrict__ w1, const float* __restrict__ b1,
                           const float* __restrict__ w2, const float* __restrict__ b2,
                           float* __restrict__ out) {
    int g = threadIdx.x;
    if (g >= 64) return;
    float inv = 1.f / fmaxf(gcnt[g], 1.f);
    float hid[32];
    #pragma unroll
    for (int j = 0; j < 32; ++j) hid[j] = b1[j];
    for (int d = 0; d < 64; ++d) {
        float v = gsum[g * 64 + d] * inv;
        #pragma unroll
        for (int j = 0; j < 32; ++j) hid[j] += v * w1[d * 32 + j];
    }
    float o = b2[0];
    #pragma unroll
    for (int j = 0; j < 32; ++j) o += fmaxf(hid[j], 0.f) * w2[j];
    out[g] = o;
}

// ---------------------------------------------------------------------------
// Launcher
// ---------------------------------------------------------------------------
extern "C" void kernel_launch(void* const* d_in, const int* in_sizes, int n_in,
                              void* d_out, int out_size, void* d_ws, size_t ws_size,
                              hipStream_t stream) {
    const float* x        = (const float*)d_in[0];
    const int*   edge_src = (const int*)  d_in[1];
    const int*   edge_dst = (const int*)  d_in[2];
    const float* edge_w   = (const float*)d_in[3];
    const int*   gids     = (const int*)  d_in[4];
    const float* w_self1  = (const float*)d_in[5];
    const float* w_rel1   = (const float*)d_in[6];
    const float* bias1    = (const float*)d_in[7];
    const float* w_q1     = (const float*)d_in[8];
    const float* w_k1     = (const float*)d_in[9];
    const float* w_att1   = (const float*)d_in[10];
    const float* w_self2  = (const float*)d_in[11];
    const float* w_rel2   = (const float*)d_in[12];
    const float* bias2    = (const float*)d_in[13];
    const float* w_q2     = (const float*)d_in[14];
    const float* w_k2     = (const float*)d_in[15];
    const float* w_att2   = (const float*)d_in[16];
    const float* mlp_w1   = (const float*)d_in[17];
    const float* mlp_b1   = (const float*)d_in[18];
    const float* mlp_w2   = (const float*)d_in[19];
    const float* mlp_b2   = (const float*)d_in[20];

    const int N = in_sizes[4];   // graph_ids has shape (N,)
    const int E = in_sizes[1];
    const size_t NF = (size_t)N;

    // Workspace layout (floats):
    //   bufA [N*128] : self1 -> (self2 | nb2)
    //   bufB [N*128] : rel1  -> h1 -> h2
    //   bufC [N*128] : nb1   -> rel2
    //   tail [4608]  : vk/vq vectors + pool accumulators
    //   abf  [N*64]  : bf16 copy of the current GEMM A matrix (N*128 ushorts)
    float* ws   = (float*)d_ws;
    float* bufA = ws;
    float* bufB = ws + NF * 128;
    float* bufC = ws + NF * 256;
    float* tail = ws + NF * 384;
    float* vk1  = tail;        float* vq1 = tail + 128;
    float* vk2  = tail + 256;  float* vq2 = tail + 320;
    float* gsum = tail + 384;                 // 64*64
    float* gcnt = tail + 384 + 64 * 64;       // 64
    unsigned short* abf = (unsigned short*)(ws + NF * 384 + 4608);

    float* outF     = (float*)d_out;
    float* ge_out   = outF;                   // 64
    float* att1_out = outF + 64;              // 2N
    float* att2_out = outF + 64 + 2 * NF;     // 2N

    const int rowTiles = (N + 15) / 16;                 // 6250
    const dim3 gemmGrid1((rowTiles + 7) / 8, 2);        // NCOLS=128
    const dim3 gemmGrid2((rowTiles + 7) / 8, 1);        // NCOLS=64
    const int spmmBlocks = (E + 7) / 8;                 // 8 edges/block
    const int attnBlocks = (N + 7) / 8;                 // 8 nodes/block

    // Attention projection folding (tiny)
    prep_vkq_kernel<<<1, 128, 0, stream>>>(w_q1, w_k1, w_att1, vk1, vq1, 128);
    prep_vkq_kernel<<<1, 64,  0, stream>>>(w_q2, w_k2, w_att2, vk2, vq2, 64);

    // ---------------- Layer 1 ----------------
    fill_zero_kernel<<<256, 256, 0, stream>>>(bufC, NF * 128);                   // nb1 = 0
    fill_zero_kernel<<<1, 256, 0, stream>>>(gsum, 64 * 64 + 64);                 // pool accum = 0
    cvt_bf16_kernel<<<512, 256, 0, stream>>>(x, abf, NF * 32);                   // x -> bf16
    gemm_bf16_wmma_kernel<128><<<gemmGrid1, 256, 0, stream>>>(abf, w_self1, bufA, N);
    gemm_bf16_wmma_kernel<128><<<gemmGrid1, 256, 0, stream>>>(abf, w_rel1,  bufB, N);
    spmm_kernel<4><<<spmmBlocks, 256, 0, stream>>>(edge_src, edge_dst, edge_w,
                                                   bufB, bufC, E);
    attn_kernel<128><<<attnBlocks, 256, 0, stream>>>(bufA, bufC, vk1, vq1, bias1,
                                                     bufB /*h1*/, att1_out, N, 1);

    // ---------------- Layer 2 ----------------
    float* self2 = bufA;
    float* nb2   = bufA + NF * 64;
    float* rel2  = bufC;
    cvt_bf16_kernel<<<512, 256, 0, stream>>>(bufB, abf, NF * 32);                // h1 -> bf16
    gemm_bf16_wmma_kernel<64><<<gemmGrid2, 256, 0, stream>>>(abf, w_self2, self2, N);
    gemm_bf16_wmma_kernel<64><<<gemmGrid2, 256, 0, stream>>>(abf, w_rel2,  rel2,  N);
    fill_zero_kernel<<<256, 256, 0, stream>>>(nb2, NF * 64);                     // nb2 = 0
    spmm_kernel<2><<<spmmBlocks, 256, 0, stream>>>(edge_src, edge_dst, edge_w,
                                                   rel2, nb2, E);
    attn_kernel<64><<<attnBlocks, 256, 0, stream>>>(self2, nb2, vk2, vq2, bias2,
                                                    bufB /*h2*/, att2_out, N, 0);

    // ---------------- Pool + MLP ----------------
    const int nodesPerBlock = 1024;
    pool_kernel<<<(N + nodesPerBlock - 1) / nodesPerBlock, 256, 0, stream>>>(
        bufB, gids, gsum, gcnt, N, nodesPerBlock);
    mlp_kernel<<<1, 64, 0, stream>>>(gsum, gcnt, mlp_w1, mlp_b1, mlp_w2, mlp_b2, ge_out);
}
